// MultiHeadAttention_52398601011223
// MI455X (gfx1250) — matrix operand order repaired and checked by reading, not compiled, not run on hardware
//
#include <hip/hip_runtime.h>
#include <hip/hip_bf16.h>
#include <math.h>

// ---------------------------------------------------------------------------
// Transformer MHA block (proj -> masked flash attention -> out proj -> LN)
// MI455X (gfx1250, wave32). All GEMMs on v_wmma_f32_16x16x32_f16 (f16 in,
// f32 accumulate). Weights are transposed+converted to f16 once so every
// WMMA B fragment is a single contiguous 32B per-lane load. K stored
// [B,H,S,64], V stored transposed [B,H,64,S]. Flash-style online softmax:
// the 2048x2048 score matrices never touch HBM.
// ---------------------------------------------------------------------------

#define DMODEL 768
#define NHEADS 12
#define DHEAD  64
#define BATCH  4
#define SEQ    2048
#define MROWS  (BATCH * SEQ)       // 8192
#define LN_EPS 1e-5f

typedef __attribute__((ext_vector_type(16))) _Float16 v16h;
typedef __attribute__((ext_vector_type(8)))  _Float16 v8h;
typedef __attribute__((ext_vector_type(8)))  float    v8f;

// --- WMMA helper ------------------------------------------------------------
__device__ __forceinline__ v8f wmma_f16(v16h a, v16h b, v8f c) {
    return __builtin_amdgcn_wmma_f32_16x16x32_f16(
        /*neg_a=*/false, a, /*neg_b=*/false, b,
        /*c_mod=*/(short)0, c, /*reuse_a=*/false, /*reuse_b=*/false);
}

// A fragment (16x32 f16). Order used here (matches B loads): lane m=lane&15, half=lane>>4;
// element e -> K = e + 16*half  (one contiguous 16-half run).
__device__ __forceinline__ v16h load_a_f16(const _Float16* rowbase, int k0, int half) {
    v8h s0 = *(const v8h*)(rowbase + k0 + 16 * half);
    v8h s1 = *(const v8h*)(rowbase + k0 + 16 * half + 8);
    v16h a;
#pragma unroll
    for (int e = 0; e < 8; ++e) { a[e] = s0[e]; a[8 + e] = s1[e]; }
    return a;
}

// A fragment from fp32 row-major source (convert to f16).
__device__ __forceinline__ v16h load_a_f32(const float* rowbase, int k0, int half) {
    const float4 f0 = *(const float4*)(rowbase + k0 + 16 * half);
    const float4 f1 = *(const float4*)(rowbase + k0 + 16 * half + 4);
    const float4 f2 = *(const float4*)(rowbase + k0 + 16 * half + 8);
    const float4 f3 = *(const float4*)(rowbase + k0 + 16 * half + 12);
    v16h a;
    a[0]  = (_Float16)f0.x; a[1]  = (_Float16)f0.y; a[2]  = (_Float16)f0.z; a[3]  = (_Float16)f0.w;
    a[4]  = (_Float16)f1.x; a[5]  = (_Float16)f1.y; a[6]  = (_Float16)f1.z; a[7]  = (_Float16)f1.w;
    a[8]  = (_Float16)f2.x; a[9]  = (_Float16)f2.y; a[10] = (_Float16)f2.z; a[11] = (_Float16)f2.w;
    a[12] = (_Float16)f3.x; a[13] = (_Float16)f3.y; a[14] = (_Float16)f3.z; a[15] = (_Float16)f3.w;
    return a;
}

// ---------------------------------------------------------------------------
// Kernel 0: weight transpose + f32->f16 convert.  Wt[n][k] = (f16) W[k][n].
// 32x32 LDS tile; grid (768/32, 768/32), 256 threads.
// ---------------------------------------------------------------------------
__global__ __launch_bounds__(256)
void wtrans_kernel(const float* __restrict__ W, _Float16* __restrict__ Wt) {
    __shared__ float tile[32][33];
    const int n0 = blockIdx.x * 32;
    const int k0 = blockIdx.y * 32;
    const int tx = threadIdx.x & 31;
    const int ty = threadIdx.x >> 5;     // 0..7
#pragma unroll
    for (int r = 0; r < 4; ++r)
        tile[ty + 8 * r][tx] = W[(size_t)(k0 + ty + 8 * r) * DMODEL + n0 + tx];
    __syncthreads();
#pragma unroll
    for (int r = 0; r < 4; ++r)
        Wt[(size_t)(n0 + ty + 8 * r) * DMODEL + k0 + tx] =
            (_Float16)tile[tx][ty + 8 * r];
}

// ---------------------------------------------------------------------------
// Kernel 1: projection GEMM  out = X @ W + bias, f16 output.
// Wt is the f16 TRANSPOSED weight [N][K] -> B fragments contiguous per lane.
// Each wave computes a 16x64 strip (4 N-tiles share one A fragment).
// mode 0: out[b][h][s][d] (q, k);  mode 1: out[b][h][d][s] (v, transposed).
// ---------------------------------------------------------------------------
__global__ __launch_bounds__(256)
void proj_kernel(const float* __restrict__ X, const _Float16* __restrict__ Wt,
                 const float* __restrict__ bias, _Float16* __restrict__ outp,
                 int mode) {
    const int lane  = threadIdx.x & 31;
    const int wave  = threadIdx.x >> 5;
    const int mlane = lane & 15;
    const int half  = lane >> 4;

    const int M16 = MROWS / 16;                 // 512
    int task = blockIdx.x * 8 + wave;           // 0 .. 6143
    int tM = task % M16;
    int tG = task / M16;                        // 0 .. 11 (64-col group)
    int row0 = tM * 16;
    int n0   = tG * 64;

    const float* arow = X + (size_t)(row0 + mlane) * DMODEL;
    const _Float16* wb = Wt + (size_t)(n0 + mlane) * DMODEL + 16 * half;

    v8f c0 = {}, c1 = {}, c2 = {}, c3 = {};
#pragma unroll 4
    for (int k0 = 0; k0 < DMODEL; k0 += 32) {
        v16h a  = load_a_f32(arow, k0, half);
        v16h b0 = *(const v16h*)(wb + k0);
        v16h b1 = *(const v16h*)(wb + k0 + (size_t)16 * DMODEL);
        v16h b2 = *(const v16h*)(wb + k0 + (size_t)32 * DMODEL);
        v16h b3 = *(const v16h*)(wb + k0 + (size_t)48 * DMODEL);
        c0 = wmma_f16(a, b0, c0);
        c1 = wmma_f16(a, b1, c1);
        c2 = wmma_f16(a, b2, c2);
        c3 = wmma_f16(a, b3, c3);
    }

#pragma unroll
    for (int j = 0; j < 4; ++j) {
        const v8f& c = (j == 0) ? c0 : (j == 1) ? c1 : (j == 2) ? c2 : c3;
        const int col = n0 + j * 16 + mlane;
        const float bv = bias[col];
        const int h  = col >> 6;
        const int dd = col & 63;
#pragma unroll
        for (int i = 0; i < 8; ++i) {
            int row = row0 + i + 8 * half;
            int bb  = row >> 11;
            int ss  = row & (SEQ - 1);
            float val = c[i] + bv;
            size_t idx = (mode == 0)
                ? ((((size_t)bb * NHEADS + h) * SEQ + ss) * DHEAD + dd)
                : ((((size_t)bb * NHEADS + h) * DHEAD + dd) * SEQ + ss);
            outp[idx] = (_Float16)val;
        }
    }
}

// ---------------------------------------------------------------------------
// Kernel 2: masked flash attention.
// grid.x = B*H (48), grid.y = SEQ/128 (16); 8 waves, each owns 16 query rows.
// ---------------------------------------------------------------------------
__global__ __launch_bounds__(256)
void attn_kernel(const _Float16* __restrict__ q, const _Float16* __restrict__ k,
                 const _Float16* __restrict__ vT, const unsigned char* __restrict__ mask,
                 _Float16* __restrict__ ctx) {
    __shared__ _Float16 lds_p[8][16][32];

    const int lane  = threadIdx.x & 31;
    const int wave  = threadIdx.x >> 5;
    const int nlane = lane & 15;
    const int half  = lane >> 4;

    const int bh  = blockIdx.x;
    const int b_  = bh / NHEADS;
    const int h_  = bh % NHEADS;
    const int qi0 = (blockIdx.y * 8 + wave) * 16;

    const _Float16* qbase = q  + (size_t)bh * SEQ * DHEAD;
    const _Float16* kbase = k  + (size_t)bh * SEQ * DHEAD;
    const _Float16* vbase = vT + (size_t)bh * DHEAD * SEQ;

    const _Float16* qrow = qbase + (size_t)(qi0 + nlane) * DHEAD;
    v16h aQ0 = load_a_f16(qrow, 0, half);
    v16h aQ1 = load_a_f16(qrow, 32, half);

    v8f acc0 = {}, acc1 = {}, acc2 = {}, acc3 = {};
    float m_run[8], l_run[8];
#pragma unroll
    for (int i = 0; i < 8; ++i) { m_run[i] = -1e30f; l_run[i] = 0.0f; }

    const float scale = 0.125f;  // 1/sqrt(64)

    for (int j0 = 0; j0 < SEQ; j0 += 32) {
        // scores: two 16x16 tiles over keys [j0, j0+32)
        v8f s0 = {}, s1 = {};
        {
            const _Float16* kr0 = kbase + (size_t)(j0 + nlane) * DHEAD;
            const _Float16* kr1 = kbase + (size_t)(j0 + 16 + nlane) * DHEAD;
            v16h b00 = *(const v16h*)(kr0 + 16 * half);
            v16h b01 = *(const v16h*)(kr0 + 32 + 16 * half);
            v16h b10 = *(const v16h*)(kr1 + 16 * half);
            v16h b11 = *(const v16h*)(kr1 + 32 + 16 * half);
            s0 = wmma_f16(aQ0, b00, s0); s0 = wmma_f16(aQ1, b01, s0);
            s1 = wmma_f16(aQ0, b10, s1); s1 = wmma_f16(aQ1, b11, s1);
        }

        // scale + mask + online softmax (rows live in 16-lane half-groups)
#pragma unroll
        for (int i = 0; i < 8; ++i) {
            int rglob = qi0 + i + 8 * half;
            size_t moff = ((size_t)b_ * SEQ + rglob) * SEQ + j0;
            float sv0 = s0[i] * scale;
            float sv1 = s1[i] * scale;
            if (mask[moff + nlane])       sv0 = -1e9f;
            if (mask[moff + 16 + nlane])  sv1 = -1e9f;

            float mx = fmaxf(sv0, sv1);
#pragma unroll
            for (int off = 8; off >= 1; off >>= 1)
                mx = fmaxf(mx, __shfl_xor(mx, off, 32));
            float mnew = fmaxf(m_run[i], mx);
            float p0 = __expf(sv0 - mnew);
            float p1 = __expf(sv1 - mnew);
            float rs = p0 + p1;
#pragma unroll
            for (int off = 8; off >= 1; off >>= 1)
                rs += __shfl_xor(rs, off, 32);
            float corr = __expf(m_run[i] - mnew);
            l_run[i] = l_run[i] * corr + rs;
            m_run[i] = mnew;
            acc0[i] *= corr; acc1[i] *= corr; acc2[i] *= corr; acc3[i] *= corr;

            lds_p[wave][i + 8 * half][nlane]      = (_Float16)p0;
            lds_p[wave][i + 8 * half][16 + nlane] = (_Float16)p1;
        }
        __syncthreads();

        v16h pa = load_a_f16(&lds_p[wave][nlane][0], 0, half);
        __syncthreads();

        // context += P @ V : V^T fragments contiguous per lane
#pragma unroll
        for (int f = 0; f < 4; ++f) {
            const _Float16* vr = vbase + (size_t)(f * 16 + nlane) * SEQ + j0 + 16 * half;
            v16h bv = *(const v16h*)vr;
            v8f* accp = (f == 0) ? &acc0 : (f == 1) ? &acc1 : (f == 2) ? &acc2 : &acc3;
            *accp = wmma_f16(pa, bv, *accp);
        }
    }

#pragma unroll
    for (int i = 0; i < 8; ++i) {
        float inv = 1.0f / l_run[i];
        int rglob = qi0 + i + 8 * half;
        size_t rowoff = ((size_t)b_ * SEQ + rglob) * DMODEL + (size_t)h_ * DHEAD;
        ctx[rowoff + 0 * 16 + nlane]  = (_Float16)(acc0[i] * inv);
        ctx[rowoff + 1 * 16 + nlane]  = (_Float16)(acc1[i] * inv);
        ctx[rowoff + 2 * 16 + nlane]  = (_Float16)(acc2[i] * inv);
        ctx[rowoff + 3 * 16 + nlane]  = (_Float16)(acc3[i] * inv);
    }
}

// ---------------------------------------------------------------------------
// Kernel 3: out = LN(ctx @ Wfc + bfc + Q) * gamma + beta.
// One block per 16-row slab; each wave owns a 96-col strip (6 tiles),
// K-loop hoisted so one A fragment feeds 6 WMMAs per step.
// Wfct is the f16 transposed weight [N][K].
// ---------------------------------------------------------------------------
__global__ __launch_bounds__(256)
void outln_kernel(const _Float16* __restrict__ ctx, const _Float16* __restrict__ Wfct,
                  const float* __restrict__ bfc, const float* __restrict__ Qres,
                  const float* __restrict__ gamma, const float* __restrict__ beta,
                  float* __restrict__ out) {
    __shared__ float tile[16][DMODEL];
    __shared__ float red1[16][16];
    __shared__ float red2[16][16];
    __shared__ float s_mu[16];
    __shared__ float s_rs[16];

    const int lane  = threadIdx.x & 31;
    const int wave  = threadIdx.x >> 5;
    const int nlane = lane & 15;
    const int half  = lane >> 4;
    const int row0  = blockIdx.x * 16;
    const int n0    = wave * 96;               // 6 tiles of 16 cols

    const _Float16* crow = ctx + (size_t)(row0 + nlane) * DMODEL;
    const _Float16* wb   = Wfct + (size_t)(n0 + nlane) * DMODEL + 16 * half;

    v8f c[6] = {{}, {}, {}, {}, {}, {}};
#pragma unroll 2
    for (int k0 = 0; k0 < DMODEL; k0 += 32) {
        v16h a = load_a_f16(crow, k0, half);
#pragma unroll
        for (int t = 0; t < 6; ++t) {
            v16h b = *(const v16h*)(wb + k0 + (size_t)(t * 16) * DMODEL);
            c[t] = wmma_f16(a, b, c[t]);
        }
    }

#pragma unroll
    for (int t = 0; t < 6; ++t) {
        int col = n0 + t * 16 + nlane;
        float bv = bfc[col];
#pragma unroll
        for (int i = 0; i < 8; ++i) {
            int r = i + 8 * half;
            tile[r][col] = c[t][i] + bv + Qres[(size_t)(row0 + r) * DMODEL + col];
        }
    }
    __syncthreads();

    // LayerNorm over 768 per row: 16 threads per row, 48 elements each.
    {
        int row = threadIdx.x >> 4;
        int seg = threadIdx.x & 15;
        float s1 = 0.0f, s2 = 0.0f;
#pragma unroll 8
        for (int cix = seg * 48; cix < seg * 48 + 48; ++cix) {
            float x = tile[row][cix];
            s1 += x;
            s2 += x * x;
        }
        red1[row][seg] = s1;
        red2[row][seg] = s2;
    }
    __syncthreads();
    if (threadIdx.x < 16) {
        int row = threadIdx.x;
        float s1 = 0.0f, s2 = 0.0f;
#pragma unroll
        for (int t = 0; t < 16; ++t) { s1 += red1[row][t]; s2 += red2[row][t]; }
        float mu  = s1 / (float)DMODEL;
        float var = s2 / (float)DMODEL - mu * mu;
        s_mu[row] = mu;
        s_rs[row] = rsqrtf(var + LN_EPS);
    }
    __syncthreads();
    {
        int row = threadIdx.x >> 4;
        int seg = threadIdx.x & 15;
        float mu = s_mu[row], rs = s_rs[row];
#pragma unroll 8
        for (int cix = seg * 48; cix < seg * 48 + 48; ++cix) {
            float x = tile[row][cix];
            out[(size_t)(row0 + row) * DMODEL + cix] =
                (x - mu) * rs * gamma[cix] + beta[cix];
        }
    }
}

// ---------------------------------------------------------------------------
// Launcher
// ---------------------------------------------------------------------------
extern "C" void kernel_launch(void* const* d_in, const int* in_sizes, int n_in,
                              void* d_out, int out_size, void* d_ws, size_t ws_size,
                              hipStream_t stream) {
    const float* Q    = (const float*)d_in[0];
    const float* K    = (const float*)d_in[1];
    const float* V    = (const float*)d_in[2];
    const unsigned char* mask = (const unsigned char*)d_in[3];
    const float* WQ   = (const float*)d_in[4];
    const float* bQ   = (const float*)d_in[5];
    const float* WK   = (const float*)d_in[6];
    const float* bK   = (const float*)d_in[7];
    const float* WV   = (const float*)d_in[8];
    const float* bV   = (const float*)d_in[9];
    const float* Wfc  = (const float*)d_in[10];
    const float* bfc  = (const float*)d_in[11];
    const float* gamma= (const float*)d_in[12];
    const float* beta = (const float*)d_in[13];
    float* out = (float*)d_out;

    // Workspace partition (all f16):
    //   q, k [B,H,S,64]; v^T [B,H,64,S]; ctx [B,S,768]; 4 transposed weights.
    const size_t nElem = (size_t)BATCH * NHEADS * SEQ * DHEAD;   // 6,291,456
    const size_t wElem = (size_t)DMODEL * DMODEL;                // 589,824
    _Float16* qws  = (_Float16*)d_ws;
    _Float16* kws  = qws + nElem;
    _Float16* vws  = kws + nElem;
    _Float16* cws  = vws + nElem;
    _Float16* wqT  = cws + nElem;
    _Float16* wkT  = wqT + wElem;
    _Float16* wvT  = wkT + wElem;
    _Float16* wfcT = wvT + wElem;

    const dim3 tgrid(DMODEL / 32, DMODEL / 32);                  // 24 x 24
    wtrans_kernel<<<tgrid, 256, 0, stream>>>(WQ,  wqT);
    wtrans_kernel<<<tgrid, 256, 0, stream>>>(WK,  wkT);
    wtrans_kernel<<<tgrid, 256, 0, stream>>>(WV,  wvT);
    wtrans_kernel<<<tgrid, 256, 0, stream>>>(Wfc, wfcT);

    const int projBlocks = (MROWS / 16) * (DMODEL / 64) / 8;     // 768
    proj_kernel<<<projBlocks, 256, 0, stream>>>(Q, wqT, bQ, qws, 0);
    proj_kernel<<<projBlocks, 256, 0, stream>>>(K, wkT, bK, kws, 0);
    proj_kernel<<<projBlocks, 256, 0, stream>>>(V, wvT, bV, vws, 1);

    attn_kernel<<<dim3(BATCH * NHEADS, SEQ / 128), 256, 0, stream>>>(
        qws, kws, vws, mask, cws);

    outln_kernel<<<MROWS / 16, 256, 0, stream>>>(
        cws, wfcT, bfc, Q, gamma, beta, out);
}
